// LSTMCharacterPredictor_28338194219184
// MI455X (gfx1250) — matrix operand order repaired and checked by reading, not compiled, not run on hardware
//
#include <hip/hip_runtime.h>

typedef __attribute__((ext_vector_type(16))) _Float16 v16h;
typedef __attribute__((ext_vector_type(8)))  _Float16 v8h;
typedef __attribute__((ext_vector_type(8)))  float    v8f;

#define T_STEPS 512
#define BATCH   64
#define EDIM    512
#define HDIM    1024
#define VOCAB   256

#define NBLOCKS 64                    // persistent grid: one block per 16 hidden units
#define WIH_STRIDE (EDIM + 8)         // padded LDS row strides (bank-conflict-free)
#define WHH_STRIDE (HDIM + 8)

// dynamic LDS layout (halves / floats)
#define WIH_LDS_HALVES  (64 * WIH_STRIDE)                 // 33280 h = 66560 B
#define WHH_LDS_HALVES  (64 * WHH_STRIDE)                 // 66048 h = 132096 B
#define GSM_FLOATS      (4 * BATCH * 16)                  // 4096 f = 16384 B
#define BIAS_FLOATS     64                                //   64 f = 256 B
#define SMEM_BYTES ((WIH_LDS_HALVES + WHH_LDS_HALVES) * 2 + (GSM_FLOATS + BIAS_FLOATS) * 4)

// ---------------- setup kernels ----------------

__global__ void f32_to_f16_kernel(const float* __restrict__ src, _Float16* __restrict__ dst, int n) {
    int i = blockIdx.x * blockDim.x + threadIdx.x;
    if (i < n) dst[i] = (_Float16)src[i];
}

__global__ void bias_combine_kernel(const float* __restrict__ b_ih, const float* __restrict__ b_hh,
                                    float* __restrict__ bias, int n) {
    int i = blockIdx.x * blockDim.x + threadIdx.x;
    if (i < n) bias[i] = b_ih[i] + b_hh[i];
}

__global__ void init_hist0_kernel(const float* __restrict__ output0, _Float16* __restrict__ h0) {
    int i = blockIdx.x * blockDim.x + threadIdx.x;   // 0..B*H-1
    h0[i] = (_Float16)output0[i & (HDIM - 1)];       // hist[0] = broadcast(output0)
}

__global__ void zero_counter_kernel(unsigned* __restrict__ bar) {
    if (threadIdx.x == 0 && blockIdx.x == 0) *bar = 0u;
}

// ---------------- persistent LSTM kernel ----------------

__device__ __forceinline__ float sigmoidf_(float x) { return 1.0f / (1.0f + __expf(-x)); }

// grid: 64 blocks x 128 threads (4 waves, one gate type each); ~210KB LDS => 1 block/WGP
__global__ void __launch_bounds__(128)
lstm_persistent_kernel(const _Float16* __restrict__ enc,    // [V, E]  f16 embeddings
                       const _Float16* __restrict__ Wih,    // [4H, E] f16
                       const _Float16* __restrict__ Whh,    // [4H, H] f16
                       const float*    __restrict__ bias,   // [4H]    b_ih + b_hh
                       const int*      __restrict__ ints,   // [T, B]
                       const float*    __restrict__ memory0,// [H]
                       _Float16*       __restrict__ hist,   // [(T+1), B, H] f16
                       unsigned*       __restrict__ bar)    // grid barrier counter (pre-zeroed)
{
    extern __shared__ char smem[];
    _Float16* wih_s  = (_Float16*)smem;                         // 64 rows x WIH_STRIDE
    _Float16* whh_s  = wih_s + WIH_LDS_HALVES;                  // 64 rows x WHH_STRIDE
    float*    gsm    = (float*)(whh_s + WHH_LDS_HALVES);        // [4][BATCH][16]
    float*    bias_s = gsm + GSM_FLOATS;                        // [4][16]

    const int lane   = threadIdx.x & 31;
    const int wave   = threadIdx.x >> 5;          // gate type: 0=i 1=f 2=g 3=o
    const int j16    = blockIdx.x;                // hidden tile (16 units)
    const int n      = lane & 15;
    const int khalfB = (lane >> 4) << 4;          // B frag: 0 or 16
    const int kqA    = (lane >> 4) << 3;          // A frag: 0 or 8
    const int hi8    = (lane >> 4) << 3;

    // ---- one-time LDS fill: this block's 64 weight rows (4 gates x 16 units) ----
    for (int idx = threadIdx.x; idx < 64 * (EDIM / 8); idx += 128) {
        int r  = idx / (EDIM / 8);
        int kk = (idx % (EDIM / 8)) * 8;
        int grow = (r >> 4) * HDIM + j16 * 16 + (r & 15);
        *(v8h*)&wih_s[r * WIH_STRIDE + kk] = *(const v8h*)(Wih + (size_t)grow * EDIM + kk);
    }
    for (int idx = threadIdx.x; idx < 64 * (HDIM / 8); idx += 128) {
        int r  = idx / (HDIM / 8);
        int kk = (idx % (HDIM / 8)) * 8;
        int grow = (r >> 4) * HDIM + j16 * 16 + (r & 15);
        *(v8h*)&whh_s[r * WHH_STRIDE + kk] = *(const v8h*)(Whh + (size_t)grow * HDIM + kk);
    }
    if (threadIdx.x < 64)
        bias_s[threadIdx.x] = bias[(threadIdx.x >> 4) * HDIM + j16 * 16 + (threadIdx.x & 15)];

    // ---- cell state lives in registers: thread owns the same 8 cells all 512 steps ----
    float creg[8];
    #pragma unroll
    for (int r = 0; r < 8; ++r) {
        int cell = threadIdx.x * 8 + r;
        creg[r] = memory0[j16 * 16 + (cell & 15)];
    }
    __syncthreads();

    const _Float16* wih_row = wih_s + (size_t)(wave * 16 + n) * WIH_STRIDE;
    const _Float16* whh_row = whh_s + (size_t)(wave * 16 + n) * WHH_STRIDE;

    for (int t = 0; t < T_STEPS; ++t) {
        const int*      ints_t = ints + t * BATCH;
        const _Float16* h_prev = hist + (size_t)t * BATCH * HDIM;
        _Float16*       h_out  = hist + (size_t)(t + 1) * BATCH * HDIM;

        // A rows: 4 batch tiles of 16; x part gathered from embeddings
        const _Float16* xrow[4];
        #pragma unroll
        for (int mt = 0; mt < 4; ++mt)
            xrow[mt] = enc + (size_t)ints_t[mt * 16 + n] * EDIM;

        v8f acc[4] = {};

        // K over E: input projection (B frags from LDS, A gathered from L2)
        #pragma unroll 2
        for (int k = 0; k < EDIM; k += 32) {
            union { v16h v; v8h h[2]; } bfu;
            bfu.h[0] = *(const v8h*)(wih_row + k + kqA);
            bfu.h[1] = *(const v8h*)(wih_row + k + 16 + kqA);
            v16h bf = bfu.v;
            #pragma unroll
            for (int mt = 0; mt < 4; ++mt) {
                union { v16h v; v8h h[2]; } a;
                a.h[0] = *(const v8h*)(xrow[mt] + k + kqA);
                a.h[1] = *(const v8h*)(xrow[mt] + k + 16 + kqA);
                acc[mt] = __builtin_amdgcn_wmma_f32_16x16x32_f16(
                    false, a.v, false, bf, (short)0, acc[mt], false, false);
            }
        }
        // K over H: recurrent projection
        #pragma unroll 2
        for (int k = 0; k < HDIM; k += 32) {
            union { v16h v; v8h h[2]; } bfu;
            bfu.h[0] = *(const v8h*)(whh_row + k + kqA);
            bfu.h[1] = *(const v8h*)(whh_row + k + 16 + kqA);
            v16h bf = bfu.v;
            #pragma unroll
            for (int mt = 0; mt < 4; ++mt) {
                const _Float16* hr = h_prev + (size_t)(mt * 16 + n) * HDIM;
                union { v16h v; v8h h[2]; } a;
                a.h[0] = *(const v8h*)(hr + k + kqA);
                a.h[1] = *(const v8h*)(hr + k + 16 + kqA);
                acc[mt] = __builtin_amdgcn_wmma_f32_16x16x32_f16(
                    false, a.v, false, bf, (short)0, acc[mt], false, false);
            }
        }

        // stage gates (C/D layout: VGPR i -> M = i + 8*(lane>=16), N = lane&15)
        #pragma unroll
        for (int mt = 0; mt < 4; ++mt)
            #pragma unroll
            for (int i = 0; i < 8; ++i)
                gsm[(wave * BATCH + mt * 16 + i + hi8) * 16 + n] = acc[mt][i];

        __syncthreads();

        // cell update: 1024 cells, 8 per thread; block owns hidden units j16*16..+15
        #pragma unroll
        for (int r = 0; r < 8; ++r) {
            int cell = threadIdx.x * 8 + r;
            int b  = cell >> 4;
            int jj = cell & 15;
            float ig = sigmoidf_(gsm[(0 * BATCH + b) * 16 + jj] + bias_s[0 * 16 + jj]);
            float fg = sigmoidf_(gsm[(1 * BATCH + b) * 16 + jj] + bias_s[1 * 16 + jj]);
            float gg = tanhf    (gsm[(2 * BATCH + b) * 16 + jj] + bias_s[2 * 16 + jj]);
            float og = sigmoidf_(gsm[(3 * BATCH + b) * 16 + jj] + bias_s[3 * 16 + jj]);
            float cc = fg * creg[r] + ig * gg;
            creg[r] = cc;
            h_out[b * HDIM + j16 * 16 + jj] = (_Float16)(og * tanhf(cc));
        }

        // device-wide generation barrier: all blocks' h_out visible before step t+1
        __syncthreads();
        if (threadIdx.x == 0) {
            __threadfence();
            __hip_atomic_fetch_add(bar, 1u, __ATOMIC_RELEASE, __HIP_MEMORY_SCOPE_AGENT);
            unsigned target = (unsigned)(t + 1) * NBLOCKS;
            while (__hip_atomic_load(bar, __ATOMIC_ACQUIRE, __HIP_MEMORY_SCOPE_AGENT) < target)
                __builtin_amdgcn_s_sleep(2);
        }
        __syncthreads();
    }
}

// ---------------- decoder GEMM: [T*B, H] x [H, V] ----------------

__global__ void __launch_bounds__(128)
decoder_kernel(const _Float16* __restrict__ hist,   // [T*B, H] f16 (= shifted outs)
               const _Float16* __restrict__ Wdec,   // [V, H]   f16
               const float*    __restrict__ b_dec,  // [V]
               float*          __restrict__ out)    // [T*B, V] f32
{
    const int lane   = threadIdx.x & 31;
    const int wave   = threadIdx.x >> 5;
    const int m0     = (blockIdx.x * 4 + wave) * 16;
    const int colbase = blockIdx.y * 16;
    const int n      = lane & 15;
    const int khalfB = (lane >> 4) << 4;
    const int kqA    = (lane >> 4) << 3;

    const _Float16* arow = hist + (size_t)(m0 + n) * HDIM;
    const _Float16* brow = Wdec + (size_t)(colbase + n) * HDIM;

    v8f acc = {};
    #pragma unroll 4
    for (int k = 0; k < HDIM; k += 32) {
        union { v16h v; v8h h[2]; } bfu;
        bfu.h[0] = *(const v8h*)(brow + k + kqA);
        bfu.h[1] = *(const v8h*)(brow + k + 16 + kqA);
        v16h bf = bfu.v;
        union { v16h v; v8h h[2]; } a;
        a.h[0] = *(const v8h*)(arow + k + kqA);
        a.h[1] = *(const v8h*)(arow + k + 16 + kqA);
        acc = __builtin_amdgcn_wmma_f32_16x16x32_f16(
            false, a.v, false, bf, (short)0, acc, false, false);
    }

    const int hi8 = (lane >> 4) << 3;
    float bb = b_dec[colbase + n];
    #pragma unroll
    for (int i = 0; i < 8; ++i)
        out[(size_t)(m0 + i + hi8) * VOCAB + colbase + n] = acc[i] + bb;
}

// ---------------- launch ----------------

extern "C" void kernel_launch(void* const* d_in, const int* in_sizes, int n_in,
                              void* d_out, int out_size, void* d_ws, size_t ws_size,
                              hipStream_t stream) {
    (void)in_sizes; (void)n_in; (void)out_size; (void)ws_size;

    const int*   d_ints = (const int*)  d_in[0];   // [T, B]
    const float* d_enc  = (const float*)d_in[1];   // [V, E]
    const float* d_out0 = (const float*)d_in[2];   // [H]
    const float* d_mem0 = (const float*)d_in[3];   // [H]
    const float* d_Wih  = (const float*)d_in[4];   // [4H, E]
    const float* d_Whh  = (const float*)d_in[5];   // [4H, H]
    const float* d_bih  = (const float*)d_in[6];   // [4H]
    const float* d_bhh  = (const float*)d_in[7];   // [4H]
    const float* d_Wdec = (const float*)d_in[8];   // [V, H]
    const float* d_bdec = (const float*)d_in[9];   // [V]
    float* out = (float*)d_out;

    // workspace carve-out (~77 MB)
    char* ws = (char*)d_ws;
    size_t off = 0;
    auto carve = [&](size_t bytes) {
        void* p = ws + off;
        off = (off + bytes + 255) & ~(size_t)255;
        return p;
    };
    _Float16* enc_h  = (_Float16*)carve((size_t)VOCAB * EDIM * 2);
    _Float16* wih_h  = (_Float16*)carve((size_t)4 * HDIM * EDIM * 2);
    _Float16* whh_h  = (_Float16*)carve((size_t)4 * HDIM * HDIM * 2);
    _Float16* wdec_h = (_Float16*)carve((size_t)VOCAB * HDIM * 2);
    float*    bias   = (float*)   carve((size_t)4 * HDIM * 4);
    unsigned* barcnt = (unsigned*)carve(256);
    _Float16* hist   = (_Float16*)carve((size_t)(T_STEPS + 1) * BATCH * HDIM * 2);

    // allow >64KB dynamic LDS for the persistent kernel (idempotent setup call)
    hipFuncSetAttribute((const void*)lstm_persistent_kernel,
                        hipFuncAttributeMaxDynamicSharedMemorySize, SMEM_BYTES);

    // precision conversion + bias fold + state init + barrier reset
    f32_to_f16_kernel<<<(VOCAB * EDIM + 255) / 256, 256, 0, stream>>>(d_enc, enc_h, VOCAB * EDIM);
    f32_to_f16_kernel<<<(4 * HDIM * EDIM + 255) / 256, 256, 0, stream>>>(d_Wih, wih_h, 4 * HDIM * EDIM);
    f32_to_f16_kernel<<<(4 * HDIM * HDIM + 255) / 256, 256, 0, stream>>>(d_Whh, whh_h, 4 * HDIM * HDIM);
    f32_to_f16_kernel<<<(VOCAB * HDIM + 255) / 256, 256, 0, stream>>>(d_Wdec, wdec_h, VOCAB * HDIM);
    bias_combine_kernel<<<(4 * HDIM + 255) / 256, 256, 0, stream>>>(d_bih, d_bhh, bias, 4 * HDIM);
    init_hist0_kernel<<<(BATCH * HDIM) / 256, 256, 0, stream>>>(d_out0, hist);
    zero_counter_kernel<<<1, 32, 0, stream>>>(barcnt);

    // one persistent kernel runs all 512 timesteps; weights LDS-resident, c in registers
    lstm_persistent_kernel<<<NBLOCKS, 128, SMEM_BYTES, stream>>>(
        enc_h, wih_h, whh_h, bias, d_ints, d_mem0, hist, barcnt);

    // hist[0..T-1] is exactly the shifted outs tensor -> one big decoder GEMM
    decoder_kernel<<<dim3((T_STEPS * BATCH) / 64, VOCAB / 16), 128, 0, stream>>>(
        hist, wdec_h, d_bdec, out);
}
